// ForceLayer_62491774156911
// MI455X (gfx1250) — matrix operand order repaired and checked by reading, not compiled, not run on hardware
//
#include <hip/hip_runtime.h>
#include <hip/hip_bf16.h>
#include <stdint.h>

#define N_NODES   10000
#define N_EDGES   160000
#define IN_DIM    128
#define HID       512
#define OUT_DIM   128
#define HEADS     16
#define HEAD_DIM  8
#define EDGE_F    64
#define TIME_F    16
#define KV_IN     320   // EDGE_F + 2*IN_DIM
#define V_IN      336   // KV_IN + TIME_F
#define V_IN_PAD  352   // padded to multiple of 32 for WMMA K
#define LN_EPS    1e-5f
#define INV_SCALE 0.3535533905932738f   // 1/sqrt(HEAD_DIM)

typedef __bf16 bf16_t;
typedef bf16_t v16bf __attribute__((ext_vector_type(16)));
typedef bf16_t v8bf  __attribute__((ext_vector_type(8)));
typedef float  v8f   __attribute__((ext_vector_type(8)));

// ---------------------------------------------------------------------------
// WMMA helpers (CDNA5 wave32, 16x16x32 bf16 -> f32)
// ---------------------------------------------------------------------------
__device__ inline v8f wmma_bf16(v16bf a, v16bf b, v8f c) {
  return __builtin_amdgcn_wmma_f32_16x16x32_bf16(
      /*neg_a=*/false, a, /*neg_b=*/false, b,
      /*c_mod=*/(short)0, c, /*reuse_a=*/false, /*reuse_b=*/false);
}

// A fragment: 16x32 bf16 tile from row-major storage (row stride lda elems).
// lanes 0-15: row M=lane, K=0..7 & 16..23; lanes 16-31: K=8..15 & 24..31.
__device__ inline v16bf load_a_frag(const bf16_t* __restrict__ base, int lda, int lane) {
  const int row = lane & 15;
  const int kh  = (lane >> 4) & 1;
  const bf16_t* p = base + row * lda + kh * 8;
  v8bf lo = *(const v8bf*)(p);
  v8bf hi = *(const v8bf*)(p + 16);
  v16bf r;
#pragma unroll
  for (int i = 0; i < 8; ++i) { r[i] = lo[i]; r[8 + i] = hi[i]; }
  return r;
}

// B fragment: 32x16 bf16 tile from PRE-TRANSPOSED weights wT[N][Kpad]
// lanes 0-15: K=k0+0..7 & k0+16..23, lanes 16-31: K=k0+8..15 & k0+24..31.
__device__ inline v16bf load_b_frag(const bf16_t* __restrict__ wT, int kpad,
                                    int n0, int k0, int lane) {
  const int col = lane & 15;
  const int kh  = (lane >> 4) & 1;
  const bf16_t* p = wT + (size_t)(n0 + col) * kpad + k0 + kh * 8;
  v8bf lo = *(const v8bf*)(p);
  v8bf hi = *(const v8bf*)(p + 16);
  v16bf r;
#pragma unroll
  for (int i = 0; i < 8; ++i) { r[i] = lo[i]; r[8 + i] = hi[i]; }
  return r;
}

__device__ inline void reduce16(float& s, float& sq) {
#pragma unroll
  for (int m = 1; m < 16; m <<= 1) {
    s  += __shfl_xor(s,  m, 16);
    sq += __shfl_xor(sq, m, 16);
  }
}

// ---------------------------------------------------------------------------
// Hidden layer for a 64-row macro tile:
//   sH[64][512] = relu(LN(sA[64][lda] @ W1 + b1))   (bf16 out)
// 8 waves; wave wv owns 64 hidden columns, all 4 row-tiles; 4x4 v8f register
// accumulators; per-row LN stats kept in registers + tiny LDS cross-wave
// combine (no fp32 activation round-trip through LDS).
// ---------------------------------------------------------------------------
template <int KCHUNKS>
__device__ inline void hidden_layer_64(
    const bf16_t* __restrict__ sA, int lda,
    const bf16_t* __restrict__ w1T, int kpad,
    const float* __restrict__ b1, const float* __restrict__ g,
    const float* __restrict__ be,
    bf16_t* __restrict__ sH,
    float* __restrict__ sPart /* [8][64][2] */,
    float* __restrict__ sMuInv /* [64][2] */,
    int tid, int lane, int wv)
{
  const int kh   = lane >> 4;
  const int cloc = lane & 15;
  const v8f vzero = {};

  v8f acc[4][4];
#pragma unroll
  for (int tt = 0; tt < 4; ++tt)
#pragma unroll
    for (int m = 0; m < 4; ++m) acc[tt][m] = vzero;

  __builtin_prefetch(w1T + (size_t)(wv * 64) * kpad, 0, 1);  // global_prefetch

  for (int kk = 0; kk < KCHUNKS; ++kk) {
    v16bf a[4];
#pragma unroll
    for (int m = 0; m < 4; ++m)
      a[m] = load_a_frag(sA + (size_t)(m * 16) * lda + kk * 32, lda, lane);
#pragma unroll
    for (int tt = 0; tt < 4; ++tt) {
      v16bf b = load_b_frag(w1T, kpad, wv * 64 + tt * 16, kk * 32, lane);
#pragma unroll
      for (int m = 0; m < 4; ++m)
        acc[tt][m] = wmma_bf16(a[m], b, acc[tt][m]);
    }
  }

  // + bias (per column)
#pragma unroll
  for (int tt = 0; tt < 4; ++tt) {
    const float bias = b1[wv * 64 + tt * 16 + cloc];
#pragma unroll
    for (int m = 0; m < 4; ++m)
#pragma unroll
      for (int j = 0; j < 8; ++j) acc[tt][m][j] += bias;
  }

  // per-row (sum, sumsq) over this wave's 64 columns -> sPart
#pragma unroll
  for (int m = 0; m < 4; ++m) {
#pragma unroll
    for (int j = 0; j < 8; ++j) {
      float s  = acc[0][m][j] + acc[1][m][j] + acc[2][m][j] + acc[3][m][j];
      float sq = acc[0][m][j] * acc[0][m][j] + acc[1][m][j] * acc[1][m][j]
               + acc[2][m][j] * acc[2][m][j] + acc[3][m][j] * acc[3][m][j];
      reduce16(s, sq);
      if (cloc == 0) {
        const int row = m * 16 + kh * 8 + j;
        sPart[(wv * 64 + row) * 2 + 0] = s;
        sPart[(wv * 64 + row) * 2 + 1] = sq;
      }
    }
  }
  __syncthreads();

  if (tid < 64) {
    float s = 0.f, sq = 0.f;
#pragma unroll
    for (int w = 0; w < 8; ++w) {
      s  += sPart[(w * 64 + tid) * 2 + 0];
      sq += sPart[(w * 64 + tid) * 2 + 1];
    }
    const float mu = s * (1.f / HID);
    sMuInv[tid * 2 + 0] = mu;
    sMuInv[tid * 2 + 1] = rsqrtf(sq * (1.f / HID) - mu * mu + LN_EPS);
  }
  __syncthreads();

  // normalize + relu + bf16 store directly from registers
  float gg[4], bb[4];
#pragma unroll
  for (int tt = 0; tt < 4; ++tt) {
    const int col = wv * 64 + tt * 16 + cloc;
    gg[tt] = g[col]; bb[tt] = be[col];
  }
#pragma unroll
  for (int m = 0; m < 4; ++m) {
#pragma unroll
    for (int j = 0; j < 8; ++j) {
      const int row  = m * 16 + kh * 8 + j;
      const float mu  = sMuInv[row * 2 + 0];
      const float inv = sMuInv[row * 2 + 1];
#pragma unroll
      for (int tt = 0; tt < 4; ++tt) {
        const float v = (acc[tt][m][j] - mu) * inv * gg[tt] + bb[tt];
        sH[(size_t)row * HID + wv * 64 + tt * 16 + cloc] = (bf16_t)fmaxf(v, 0.f);
      }
    }
  }
}

// Second GEMM: out[64][128] tiles; wave wv owns 16 output cols, all 4 row
// tiles; B fragment reused across the 4 row tiles.
__device__ inline void gemm2_64(const bf16_t* __restrict__ sH,
                                const bf16_t* __restrict__ w2T,
                                int lane, int wv, v8f acc[4]) {
  const int n0 = wv * 16;
  const v8f vzero = {};
#pragma unroll
  for (int m = 0; m < 4; ++m) acc[m] = vzero;
  for (int kk = 0; kk < 16; ++kk) {
    v16bf b = load_b_frag(w2T, HID, n0, kk * 32, lane);
#pragma unroll
    for (int m = 0; m < 4; ++m) {
      v16bf a = load_a_frag(sH + (size_t)(m * 16) * HID + kk * 32, HID, lane);
      acc[m] = wmma_bf16(a, b, acc[m]);
    }
  }
}

__device__ inline void atomicMaxFloat(float* addr, float val) {
  if (val >= 0.f) atomicMax((int*)addr, __float_as_int(val));
  else            atomicMin((unsigned int*)addr, __float_as_uint(val));
}

// ---------------------------------------------------------------------------
// Prep kernels
// ---------------------------------------------------------------------------
__global__ void cvt_bf16_kernel(const float* __restrict__ src,
                                bf16_t* __restrict__ dst, size_t n) {
  size_t i = (size_t)blockIdx.x * blockDim.x + threadIdx.x;
  size_t stride = (size_t)gridDim.x * blockDim.x;
  for (; i < n; i += stride) dst[i] = (bf16_t)src[i];
}

// w[K][Nc] fp32 -> wT[Nc][Kpad] bf16 (zero padded beyond K)
__global__ void transpose_w_kernel(const float* __restrict__ w,
                                   bf16_t* __restrict__ wT,
                                   int K, int Kpad, int Nc) {
  size_t total = (size_t)Nc * Kpad;
  size_t i = (size_t)blockIdx.x * blockDim.x + threadIdx.x;
  size_t stride = (size_t)gridDim.x * blockDim.x;
  for (; i < total; i += stride) {
    int n = (int)(i / Kpad), k = (int)(i % Kpad);
    wT[i] = (k < K) ? (bf16_t)w[(size_t)k * Nc + n] : (bf16_t)0.f;
  }
}

__global__ void fill_bits_kernel(unsigned int* __restrict__ p, unsigned int v, size_t n) {
  size_t i = (size_t)blockIdx.x * blockDim.x + threadIdx.x;
  size_t stride = (size_t)gridDim.x * blockDim.x;
  for (; i < n; i += stride) p[i] = v;
}

// ---------------------------------------------------------------------------
// Node q-MLP: q = MLP(h)  (N x 128 -> N x 128), 64-node macro tiles
// ---------------------------------------------------------------------------
__global__ __launch_bounds__(256) void q_mlp_kernel(
    const float* __restrict__ h, const bf16_t* __restrict__ w1T,
    const float* __restrict__ b1, const float* __restrict__ g,
    const float* __restrict__ be, const bf16_t* __restrict__ w2T,
    const float* __restrict__ b2, float* __restrict__ qout) {
  __shared__ __align__(16) bf16_t sA[64 * IN_DIM];
  __shared__ __align__(16) bf16_t sH[64 * HID];
  __shared__ __align__(16) float  sPart[8 * 64 * 2];
  __shared__ __align__(16) float  sMuInv[64 * 2];

  const int tid   = threadIdx.x;
  const int lane  = tid & 31;
  const int wv    = tid >> 5;
  const int node0 = blockIdx.x * 64;

  for (int i = tid; i < 64 * IN_DIM; i += 256) {
    const int r = i >> 7, c = i & 127;
    const int node = node0 + r;
    sA[i] = (node < N_NODES) ? (bf16_t)h[(size_t)node * IN_DIM + c] : (bf16_t)0.f;
  }
  __syncthreads();

  hidden_layer_64<4>(sA, IN_DIM, w1T, IN_DIM, b1, g, be, sH, sPart, sMuInv,
                     tid, lane, wv);   // K = 128
  __syncthreads();

  v8f acc[4];
  gemm2_64(sH, w2T, lane, wv, acc);    // K = 512, 16 out cols per wave
  {
    const int cloc = lane & 15;
    const int kh   = lane >> 4;
    const int col  = wv * 16 + cloc;
    const float bias = b2[col];
#pragma unroll
    for (int m = 0; m < 4; ++m)
#pragma unroll
      for (int j = 0; j < 8; ++j) {
        const int node = node0 + m * 16 + kh * 8 + j;
        if (node < N_NODES)
          qout[(size_t)node * OUT_DIM + col] = acc[m][j] + bias;
      }
  }
}

// ---------------------------------------------------------------------------
// Edge kernel: gather kv, k-MLP, v-MLP, logits + segment max (64 edges/block)
// ---------------------------------------------------------------------------
struct EdgeParams {
  const bf16_t* efbf;  // E x 64
  const bf16_t* hbf;   // N x 128
  const bf16_t* tbf;   // N x 16
  const int* src; const int* dst;
  const bf16_t* wk1T; const float* kb1; const float* kg; const float* kbe;
  const bf16_t* wk2T; const float* kb2;
  const bf16_t* wv1T; const float* vb1; const float* vg; const float* vbe;
  const bf16_t* wv2T; const float* vb2;
  const float* q;      // N x 128 (from q_mlp)
  float* logits;       // E x 16
  float* vout;         // E x 16
  float* mx;           // N x 16 (init -inf)
};

__global__ __launch_bounds__(256) void edge_mlp_kernel(EdgeParams P) {
  __shared__ __align__(16) bf16_t sA[64 * V_IN_PAD];   // 45 KB
  __shared__ __align__(16) bf16_t sH[64 * HID];        // 64 KB
  __shared__ __align__(16) float  sK[64 * OUT_DIM];    // 32 KB
  __shared__ __align__(16) float  sV[64 * HEADS];      //  4 KB
  __shared__ __align__(16) float  sPart[8 * 64 * 2];   //  4 KB
  __shared__ __align__(16) float  sMuInv[64 * 2];
  __shared__ int sSrc[64], sDst[64];

  const int tid  = threadIdx.x;
  const int lane = tid & 31;
  const int wv   = tid >> 5;
  const int e0   = blockIdx.x * 64;

  if (tid < 64) { sSrc[tid] = P.src[e0 + tid]; sDst[tid] = P.dst[e0 + tid]; }
  __syncthreads();

  // Build kv tile: [edge_feat | h[dst] | h[src] | t[dst] | zero pad]
  for (int i = tid; i < 64 * V_IN_PAD; i += 256) {
    const int r = i / V_IN_PAD, c = i % V_IN_PAD;
    bf16_t v;
    if      (c < 64)  v = P.efbf[(size_t)(e0 + r) * EDGE_F + c];
    else if (c < 192) v = P.hbf[(size_t)sDst[r] * IN_DIM + (c - 64)];
    else if (c < 320) v = P.hbf[(size_t)sSrc[r] * IN_DIM + (c - 192)];
    else if (c < 336) v = P.tbf[(size_t)sDst[r] * TIME_F + (c - 320)];
    else              v = (bf16_t)0.f;
    sA[i] = v;
  }
  __syncthreads();

  // ---- k-MLP ----
  hidden_layer_64<10>(sA, V_IN_PAD, P.wk1T, KV_IN, P.kb1, P.kg, P.kbe,
                      sH, sPart, sMuInv, tid, lane, wv);   // K = 320
  __syncthreads();
  {
    v8f acc[4];
    gemm2_64(sH, P.wk2T, lane, wv, acc);
    const int cloc = lane & 15;
    const int kh   = lane >> 4;
    const int col  = wv * 16 + cloc;
    const float bias = P.kb2[col];
#pragma unroll
    for (int m = 0; m < 4; ++m)
#pragma unroll
      for (int j = 0; j < 8; ++j)
        sK[(size_t)(m * 16 + kh * 8 + j) * OUT_DIM + col] = acc[m][j] + bias;
  }
  __syncthreads();

  // ---- v-MLP ----
  hidden_layer_64<11>(sA, V_IN_PAD, P.wv1T, V_IN_PAD, P.vb1, P.vg, P.vbe,
                      sH, sPart, sMuInv, tid, lane, wv);   // K = 352 (padded)
  __syncthreads();
  if (wv < 4) {   // one 16x16 tile per wave, 4 row tiles total (HEADS = 16)
    const int m = wv;
    const v8f vzero = {};
    v8f acc = vzero;
    for (int kk = 0; kk < 16; ++kk) {
      v16bf b = load_b_frag(P.wv2T, HID, 0, kk * 32, lane);
      v16bf a = load_a_frag(sH + (size_t)(m * 16) * HID + kk * 32, HID, lane);
      acc = wmma_bf16(a, b, acc);
    }
    const int cloc = lane & 15;
    const int kh   = lane >> 4;
    const float bias = P.vb2[cloc];
#pragma unroll
    for (int j = 0; j < 8; ++j)
      sV[(size_t)(m * 16 + kh * 8 + j) * HEADS + cloc] = acc[j] + bias;
  }
  __syncthreads();

  // ---- logits = (q[dst] . k) / sqrt(d), segment max via float atomics ----
#pragma unroll
  for (int mm = 0; mm < 4; ++mm) {
    const int r  = mm * 16 + (tid >> 4);   // edge within tile (0..63)
    const int hd = tid & 15;               // head
    const int d  = sDst[r];
    const float* qrow = P.q + (size_t)d * OUT_DIM + hd * HEAD_DIM;
    const float* krow = sK + (size_t)r * OUT_DIM + hd * HEAD_DIM;
    float dot = 0.f;
#pragma unroll
    for (int j = 0; j < HEAD_DIM; ++j) dot += qrow[j] * krow[j];
    const float lg = dot * INV_SCALE;
    P.logits[(size_t)(e0 + r) * HEADS + hd] = lg;
    P.vout[(size_t)(e0 + r) * HEADS + hd]   = sV[(size_t)r * HEADS + hd];
    atomicMaxFloat(&P.mx[(size_t)d * HEADS + hd], lg);
  }
}

// ---------------------------------------------------------------------------
// Softmax numerator + denominators
// ---------------------------------------------------------------------------
__global__ __launch_bounds__(256) void softmax_num_kernel(
    float* __restrict__ logits /* in: logits, out: exp */,
    const float* __restrict__ mx, const int* __restrict__ dst,
    float* __restrict__ denom) {
  int e = blockIdx.x * blockDim.x + threadIdx.x;
  if (e >= N_EDGES) return;
  const int d = dst[e];
#pragma unroll
  for (int hd = 0; hd < HEADS; ++hd) {
    float ex = __expf(logits[(size_t)e * HEADS + hd] - mx[(size_t)d * HEADS + hd]);
    logits[(size_t)e * HEADS + hd] = ex;
    atomicAdd(&denom[(size_t)d * HEADS + hd], ex);
  }
}

// ---------------------------------------------------------------------------
// Weighted scatter: m = alpha * v * rel_x ; mean over heads; masked segsum
// ---------------------------------------------------------------------------
__global__ __launch_bounds__(256) void scatter_kernel(
    const float* __restrict__ ex, const float* __restrict__ denom,
    const float* __restrict__ vout, const float* __restrict__ rel_x,
    const int* __restrict__ dst, const unsigned char* __restrict__ mask,
    float* __restrict__ out /* [inner 3N | outer 3N] */) {
  int e = blockIdx.x * blockDim.x + threadIdx.x;
  if (e >= N_EDGES) return;
  const int d = dst[e];
  float s = 0.f;
#pragma unroll
  for (int hd = 0; hd < HEADS; ++hd) {
    float alpha = ex[(size_t)e * HEADS + hd] / denom[(size_t)d * HEADS + hd];
    s += alpha * vout[(size_t)e * HEADS + hd];
  }
  s *= (1.f / HEADS);  // mean over heads
  float* base = out + (mask[e] ? 0 : (size_t)3 * N_NODES) + (size_t)d * 3;
#pragma unroll
  for (int c = 0; c < 3; ++c)
    atomicAdd(&base[c], s * rel_x[(size_t)e * 3 + c]);
}

// ---------------------------------------------------------------------------
// Host launcher
// ---------------------------------------------------------------------------
extern "C" void kernel_launch(void* const* d_in, const int* in_sizes, int n_in,
                              void* d_out, int out_size, void* d_ws, size_t ws_size,
                              hipStream_t stream) {
  (void)in_sizes; (void)n_in; (void)out_size; (void)ws_size;

  const float* h    = (const float*)d_in[0];
  const float* relx = (const float*)d_in[1];
  const float* ef   = (const float*)d_in[2];
  const float* t    = (const float*)d_in[3];
  const float* xk_w1 = (const float*)d_in[4];
  const float* xk_b1 = (const float*)d_in[5];
  const float* xk_g  = (const float*)d_in[6];
  const float* xk_be = (const float*)d_in[7];
  const float* xk_w2 = (const float*)d_in[8];
  const float* xk_b2 = (const float*)d_in[9];
  const float* xv_w1 = (const float*)d_in[10];
  const float* xv_b1 = (const float*)d_in[11];
  const float* xv_g  = (const float*)d_in[12];
  const float* xv_be = (const float*)d_in[13];
  const float* xv_w2 = (const float*)d_in[14];
  const float* xv_b2 = (const float*)d_in[15];
  const float* xq_w1 = (const float*)d_in[16];
  const float* xq_b1 = (const float*)d_in[17];
  const float* xq_g  = (const float*)d_in[18];
  const float* xq_be = (const float*)d_in[19];
  const float* xq_w2 = (const float*)d_in[20];
  const float* xq_b2 = (const float*)d_in[21];
  const int*   eidx  = (const int*)d_in[22];
  const unsigned char* mask = (const unsigned char*)d_in[23];
  const int* src = eidx;
  const int* dst = eidx + N_EDGES;
  float* out = (float*)d_out;

  // carve workspace (256B-aligned slabs)
  char* wp = (char*)d_ws;
  auto take = [&](size_t bytes) -> void* {
    void* p = (void*)wp;
    wp += (bytes + 255) & ~(size_t)255;
    return p;
  };
  bf16_t* hbf  = (bf16_t*)take((size_t)N_NODES * IN_DIM * 2);
  bf16_t* efbf = (bf16_t*)take((size_t)N_EDGES * EDGE_F * 2);
  bf16_t* tbf  = (bf16_t*)take((size_t)N_NODES * TIME_F * 2);
  bf16_t* wq1T = (bf16_t*)take((size_t)HID * IN_DIM * 2);
  bf16_t* wq2T = (bf16_t*)take((size_t)OUT_DIM * HID * 2);
  bf16_t* wk1T = (bf16_t*)take((size_t)HID * KV_IN * 2);
  bf16_t* wk2T = (bf16_t*)take((size_t)OUT_DIM * HID * 2);
  bf16_t* wv1T = (bf16_t*)take((size_t)HID * V_IN_PAD * 2);
  bf16_t* wv2T = (bf16_t*)take((size_t)HEADS * HID * 2);
  float* qbuf   = (float*)take((size_t)N_NODES * OUT_DIM * 4);
  float* logits = (float*)take((size_t)N_EDGES * HEADS * 4);
  float* vbuf   = (float*)take((size_t)N_EDGES * HEADS * 4);
  float* mx     = (float*)take((size_t)N_NODES * HEADS * 4);
  float* denom  = (float*)take((size_t)N_NODES * HEADS * 4);

  auto nblk = [](size_t n) { return (unsigned)((n + 255) / 256); };

  // 1) bf16 conversions
  cvt_bf16_kernel<<<nblk((size_t)N_NODES * IN_DIM), 256, 0, stream>>>(h, hbf, (size_t)N_NODES * IN_DIM);
  cvt_bf16_kernel<<<nblk((size_t)N_EDGES * EDGE_F), 256, 0, stream>>>(ef, efbf, (size_t)N_EDGES * EDGE_F);
  cvt_bf16_kernel<<<nblk((size_t)N_NODES * TIME_F), 256, 0, stream>>>(t, tbf, (size_t)N_NODES * TIME_F);

  // 2) weight transpose + convert: w[K][Nc] -> wT[Nc][Kpad] bf16
  transpose_w_kernel<<<nblk((size_t)HID * IN_DIM), 256, 0, stream>>>(xq_w1, wq1T, IN_DIM, IN_DIM, HID);
  transpose_w_kernel<<<nblk((size_t)OUT_DIM * HID), 256, 0, stream>>>(xq_w2, wq2T, HID, HID, OUT_DIM);
  transpose_w_kernel<<<nblk((size_t)HID * KV_IN), 256, 0, stream>>>(xk_w1, wk1T, KV_IN, KV_IN, HID);
  transpose_w_kernel<<<nblk((size_t)OUT_DIM * HID), 256, 0, stream>>>(xk_w2, wk2T, HID, HID, OUT_DIM);
  transpose_w_kernel<<<nblk((size_t)HID * V_IN_PAD), 256, 0, stream>>>(xv_w1, wv1T, V_IN, V_IN_PAD, HID);
  transpose_w_kernel<<<nblk((size_t)HEADS * HID), 256, 0, stream>>>(xv_w2, wv2T, HID, HID, HEADS);

  // 3) init accumulators: mx = -inf, denom = 0, out = 0
  fill_bits_kernel<<<nblk((size_t)N_NODES * HEADS), 256, 0, stream>>>((unsigned int*)mx, 0xFF800000u, (size_t)N_NODES * HEADS);
  fill_bits_kernel<<<nblk((size_t)N_NODES * HEADS), 256, 0, stream>>>((unsigned int*)denom, 0u, (size_t)N_NODES * HEADS);
  fill_bits_kernel<<<nblk((size_t)6 * N_NODES), 256, 0, stream>>>((unsigned int*)out, 0u, (size_t)6 * N_NODES);

  // 4) q-MLP over nodes (64-node macro tiles)
  q_mlp_kernel<<<(N_NODES + 63) / 64, 256, 0, stream>>>(h, wq1T, xq_b1, xq_g, xq_be, wq2T, xq_b2, qbuf);

  // 5) per-edge k/v MLPs + logits + segment max (64 edges per block)
  EdgeParams P;
  P.efbf = efbf; P.hbf = hbf; P.tbf = tbf; P.src = src; P.dst = dst;
  P.wk1T = wk1T; P.kb1 = xk_b1; P.kg = xk_g; P.kbe = xk_be; P.wk2T = wk2T; P.kb2 = xk_b2;
  P.wv1T = wv1T; P.vb1 = xv_b1; P.vg = xv_g; P.vbe = xv_be; P.wv2T = wv2T; P.vb2 = xv_b2;
  P.q = qbuf; P.logits = logits; P.vout = vbuf; P.mx = mx;
  edge_mlp_kernel<<<N_EDGES / 64, 256, 0, stream>>>(P);

  // 6) softmax numerators + denominators
  softmax_num_kernel<<<nblk(N_EDGES), 256, 0, stream>>>(logits, mx, dst, denom);

  // 7) masked weighted scatter into inner/outer forces
  scatter_kernel<<<nblk(N_EDGES), 256, 0, stream>>>(logits, denom, vbuf, relx, dst, mask, out);
}
